// GraphNet_738734375487
// MI455X (gfx1250) — matrix operand order repaired and checked by reading, not compiled, not run on hardware
//
#include <hip/hip_runtime.h>
#include <hip/hip_bf16.h>

typedef __attribute__((ext_vector_type(16))) _Float16 v16h;
typedef __attribute__((ext_vector_type(8)))  float    v8f;

#define NN 50000
#define NE 800000
#define CH 16
#define FEF 8
#define NG 64

// ---------------------------------------------------------------- utilities
__global__ void zero_kernel(float* __restrict__ p, int n) {
    int t = blockIdx.x * blockDim.x + threadIdx.x;
    if (t < n) p[t] = 0.0f;
}

__global__ void count_kernel(const int* __restrict__ dst, float* __restrict__ cnt, int E) {
    int t = blockIdx.x * blockDim.x + threadIdx.x;
    if (t < E) atomicAdd(&cnt[dst[t]], 1.0f);
}

// ------------------------------------------------- WMMA B-fragment builder
// B is a 16x16 f32 row-major matrix W[i][o] = W[i*16+o], zero-padded to K=32.
// Layout (v_wmma_f32_16x16x32_f16 B, 32x16): lane half h, elem e<8 <-> K=8h+e;
// elems 8..15 <-> K=16+8h+(e-8) -> all zero here (K pad).
__device__ __forceinline__ v16h make_bfrag(const float* __restrict__ W, int lane) {
    v16h b;
    int col = lane & 15;
    int kb = (lane & 16) ? 8 : 0;
#pragma unroll
    for (int i = 0; i < 8; ++i) b[i] = (_Float16)W[(kb + i) * 16 + col];
#pragma unroll
    for (int i = 8; i < 16; ++i) b[i] = (_Float16)0.0f;
    return b;
}

// ------------------------------------------------------- node transform
// For a 16-node tile: A = hin[m0..m0+15, 0..15] (f16, K padded to 32).
// Emits tn[n, f, o] = (hin @ weR_f)[n,o]  f=0..7
//        tn[n, 8, o] = (hin @ be_mat)[n,o]
//        hroot[n,o]  = (hin @ root)[n,o]
__global__ void __launch_bounds__(256)
node_transform_kernel(const float* __restrict__ hin,
                      const float* __restrict__ we,   // [8,256]
                      const float* __restrict__ be,   // [256]
                      const float* __restrict__ root, // [256]
                      float* __restrict__ tn,         // [N,9,16]
                      float* __restrict__ hroot,      // [N,16]
                      int n_nodes) {
    const int lane   = threadIdx.x & 31;
    const int wave   = blockIdx.x * (blockDim.x >> 5) + (threadIdx.x >> 5);
    const int nwaves = gridDim.x * (blockDim.x >> 5);

    // Hoist all 10 B fragments out of the tile loop (80 VGPRs).
    v16h bf[10];
#pragma unroll
    for (int f = 0; f < 8; ++f) bf[f] = make_bfrag(we + f * 256, lane);
    bf[8] = make_bfrag(be, lane);
    bf[9] = make_bfrag(root, lane);

    const int ntiles = n_nodes >> 4;          // 50000/16 = 3125 exact
    const int col = lane & 15;
    const int hi  = (lane & 16) ? 8 : 0;      // C/D: M = r + hi
    const int row = lane & 15;                // A: row = lane%16
    const int kb  = (lane & 16) ? 8 : 0;      // A: K base

    for (int tile = wave; tile < ntiles; tile += nwaves) {
        const int m0 = tile << 4;

        // A fragment: a[0..7] = hin[m0+row][kb..kb+7]; a[8..15] = 0 (K pad)
        v16h a;
        const float* xr = hin + (size_t)(m0 + row) * CH + kb;
#pragma unroll
        for (int i = 0; i < 8; ++i)  a[i] = (_Float16)xr[i];
#pragma unroll
        for (int i = 8; i < 16; ++i) a[i] = (_Float16)0.0f;

        // 9 GEMMs -> tn
#pragma unroll
        for (int f = 0; f < 9; ++f) {
            v8f c = {};
            c = __builtin_amdgcn_wmma_f32_16x16x32_f16(
                    false, a, false, bf[f], (short)0, c, false, false);
            float* out = tn + (size_t)m0 * 144 + f * 16;
#pragma unroll
            for (int r = 0; r < 8; ++r)
                out[(size_t)(hi + r) * 144 + col] = c[r];
        }
        // root GEMM -> hroot
        {
            v8f c = {};
            c = __builtin_amdgcn_wmma_f32_16x16x32_f16(
                    false, a, false, bf[9], (short)0, c, false, false);
            float* out = hroot + (size_t)m0 * CH;
#pragma unroll
            for (int r = 0; r < 8; ++r)
                out[(size_t)(hi + r) * CH + col] = c[r];
        }
    }
}

// ------------------------------------------------------- edge message + scatter
// thread t = e*16 + o:  msg = tn[src,8,o] + sum_f ea[e,f]*tn[src,f,o]
__global__ void __launch_bounds__(256)
edge_message_kernel(const int* __restrict__ src, const int* __restrict__ dst,
                    const float* __restrict__ ea, const float* __restrict__ tn,
                    float* __restrict__ agg, int E) {
    int t = blockIdx.x * blockDim.x + threadIdx.x;
    int e = t >> 4, o = t & 15;
    if (e >= E) return;
    const float* tns = tn + (size_t)src[e] * 144;
    const float* eav = ea + (size_t)e * FEF;
    float m = tns[128 + o];                       // bias-matrix term
#pragma unroll
    for (int f = 0; f < 8; ++f) m += eav[f] * tns[f * 16 + o];
    atomicAdd(&agg[(size_t)dst[e] * CH + o], m);
}

// ------------------------------------------------------- finalize: root + mean-agg + bias
__global__ void finalize_kernel(const float* __restrict__ hroot,
                                const float* __restrict__ agg,
                                const float* __restrict__ cnt,
                                const float* __restrict__ bias,
                                float* __restrict__ hout, int n_nodes) {
    int t = blockIdx.x * blockDim.x + threadIdx.x;
    if (t >= n_nodes * CH) return;
    int node = t >> 4, o = t & 15;
    float c = cnt[node];
    c = c > 1.0f ? c : 1.0f;
    hout[t] = hroot[t] + agg[t] / c + bias[o];
}

// ------------------------------------------------------- global mean pool
__global__ void pool_acc_kernel(const float* __restrict__ h, const int* __restrict__ batch,
                                float* __restrict__ psum, float* __restrict__ pcnt, int n_nodes) {
    int t = blockIdx.x * blockDim.x + threadIdx.x;
    if (t >= n_nodes * CH) return;
    int node = t >> 4, o = t & 15;
    int g = batch[node];
    atomicAdd(&psum[g * CH + o], h[t]);
    if (o == 0) atomicAdd(&pcnt[g], 1.0f);
}

__global__ void pool_div_kernel(const float* __restrict__ psum, const float* __restrict__ pcnt,
                                float* __restrict__ out) {
    int t = blockIdx.x * blockDim.x + threadIdx.x;
    if (t < NG * CH) {
        float c = pcnt[t >> 4];
        c = c > 1.0f ? c : 1.0f;
        out[t] = psum[t] / c;
    }
}

// ---------------------------------------------------------------- launcher
extern "C" void kernel_launch(void* const* d_in, const int* in_sizes, int n_in,
                              void* d_out, int out_size, void* d_ws, size_t ws_size,
                              hipStream_t stream) {
    const float* x     = (const float*)d_in[0];
    const int*   ei    = (const int*)  d_in[1];   // [2, NE]
    const float* ea    = (const float*)d_in[2];   // [NE, 8]
    const int*   batch = (const int*)  d_in[3];   // [NN]
    const float* we1   = (const float*)d_in[4];
    const float* be1   = (const float*)d_in[5];
    const float* root1 = (const float*)d_in[6];
    const float* bias1 = (const float*)d_in[7];
    const float* we2   = (const float*)d_in[8];
    const float* be2   = (const float*)d_in[9];
    const float* root2 = (const float*)d_in[10];
    const float* bias2 = (const float*)d_in[11];

    const int* src = ei;
    const int* dst = ei + NE;

    // workspace layout (floats): 209*NN + 1088 ~ 42 MB
    float* ws    = (float*)d_ws;
    float* cnt   = ws;                         // NN
    float* agg   = cnt   + NN;                 // 16*NN
    float* hroot = agg   + 16 * NN;            // 16*NN
    float* h1    = hroot + 16 * NN;            // 16*NN
    float* h2    = h1    + 16 * NN;            // 16*NN
    float* tn    = h2    + 16 * NN;            // 144*NN
    float* pool  = tn    + 144 * (size_t)NN;   // 1024 psum + 64 pcnt

    const int TB = 256;
    const int nt_blocks = ( (NN / 16) + 7 ) / 8;        // 1 wave per 16-node tile
    const int nc_blocks = (NN * CH + TB - 1) / TB;
    const int e_blocks  = (NE * CH + TB - 1) / TB;

    // shared prep: edge-count histogram, pool accumulators
    zero_kernel<<<(NN + TB - 1) / TB, TB, 0, stream>>>(cnt, NN);
    zero_kernel<<<(1088 + TB - 1) / TB, TB, 0, stream>>>(pool, 1088);
    count_kernel<<<(NE + TB - 1) / TB, TB, 0, stream>>>(dst, cnt, NE);

    // ---- layer 1
    node_transform_kernel<<<nt_blocks, TB, 0, stream>>>(x, we1, be1, root1, tn, hroot, NN);
    zero_kernel<<<nc_blocks, TB, 0, stream>>>(agg, NN * CH);
    edge_message_kernel<<<e_blocks, TB, 0, stream>>>(src, dst, ea, tn, agg, NE);
    finalize_kernel<<<nc_blocks, TB, 0, stream>>>(hroot, agg, cnt, bias1, h1, NN);

    // ---- layer 2
    node_transform_kernel<<<nt_blocks, TB, 0, stream>>>(h1, we2, be2, root2, tn, hroot, NN);
    zero_kernel<<<nc_blocks, TB, 0, stream>>>(agg, NN * CH);
    edge_message_kernel<<<e_blocks, TB, 0, stream>>>(src, dst, ea, tn, agg, NE);
    finalize_kernel<<<nc_blocks, TB, 0, stream>>>(hroot, agg, cnt, bias2, h2, NN);

    // ---- global mean pool -> [64*16]
    pool_acc_kernel<<<nc_blocks, TB, 0, stream>>>(h2, batch, pool, pool + NG * CH, NN);
    pool_div_kernel<<<(NG * CH + TB - 1) / TB, TB, 0, stream>>>(pool, pool + NG * CH, (float*)d_out);
}
